// ContextualLSTM_5849745457564
// MI455X (gfx1250) — compile-verified, matrix operand order checked by reading, not run on hardware
//
#include <hip/hip_runtime.h>
#include <hip/hip_bf16.h>

// ---------------------------------------------------------------------------
// Bidirectional LSTM for MI455X (gfx1250): WMMA f16 GEMMs + persistent-WG scan
// with Tensor-Data-Mover double-buffered xg prefetch into LDS.
// ---------------------------------------------------------------------------

typedef __attribute__((ext_vector_type(16))) _Float16 v16h;
typedef __attribute__((ext_vector_type(8)))  float    v8f;
typedef __attribute__((ext_vector_type(4)))  unsigned int u32x4;
typedef __attribute__((ext_vector_type(8)))  int          i32x8;
typedef __attribute__((ext_vector_type(4)))  int          i32x4;

union V16H { v16h v; uint4 q[2]; };
union V8H  { _Float16 h[8]; uint4 q; };

#define V_   50000
#define E_   300
#define EP_  320        // E padded to multiple of 32
#define B_   32
#define T_   512
#define H_   256
#define G4_  1024       // 4*H
#define OUT_ 256
#define KT_IH 10        // 320/32
#define KT_HH 8         // 256/32
#define XSLICE_ (B_ * G4_)            // halves per timestep slice = 32768
#define XBUF_BYTES (2 * XSLICE_ * 2)  // double buffer, f16 = 131072 B

#if defined(__has_builtin)
#if __has_builtin(__builtin_amdgcn_tensor_load_to_lds) && __has_builtin(__builtin_amdgcn_s_wait_tensorcnt)
#define USE_TDM 1
#endif
#endif
#ifndef USE_TDM
#define USE_TDM 0
#endif

__device__ __forceinline__ v8f wmma_f16(v16h a, v16h b, v8f c) {
    return __builtin_amdgcn_wmma_f32_16x16x32_f16(false, a, false, b, (short)0, c, false, false);
}

__device__ __forceinline__ float sigmoidf_(float x) {
    return 1.0f / (1.0f + __expf(-x));
}

// Map half-index h (0..15) within a lane's A/B fragment to its K offset.
// lanes 0-15 hold K {0..7,16..23}; lanes 16-31 hold K {8..15,24..31}.
__device__ __forceinline__ int kOfHalf(int lane, int h) {
    int hi = lane >> 4;
    return (h < 8) ? (hi ? h + 8 : h) : (hi ? h + 16 : h + 8);
}

#if USE_TDM
// 1-D TDM descriptor: copy nhalves f16 elements global -> LDS (contiguous).
__device__ __forceinline__ void tdm_load_1d(unsigned lds_off, const void* gaddr,
                                            unsigned nhalves) {
    unsigned long long ga = (unsigned long long)(uintptr_t)gaddr;
    u32x4 g0;
    g0[0] = 1u;                                   // count=1, user descriptor
    g0[1] = lds_off;                              // lds_addr (bytes)
    g0[2] = (unsigned)(ga & 0xFFFFFFFFu);         // global_addr[31:0]
    g0[3] = (unsigned)((ga >> 32) & 0x01FFFFFFu)  // global_addr[56:32]
            | (2u << 30);                         // type=2 (image)
    i32x8 g1;
    g1[0] = (int)(1u << 16);                      // data_size=1 (2 bytes)
    g1[1] = (int)((nhalves & 0xFFFFu) << 16);     // tensor_dim0[15:0]
    g1[2] = (int)(((nhalves >> 16) & 0xFFFFu) | (1u << 16)); // dim0 hi | tensor_dim1=1
    g1[3] = (int)((nhalves & 0xFFFFu) << 16);     // tile_dim0 (16-bit)
    g1[4] = 0;                                    // tile_dim1=0, tile_dim2=0 (1-D)
    g1[5] = (int)nhalves;                         // tensor_dim0_stride lo
    g1[6] = 0;
    g1[7] = 0;
    i32x4 z4 = {0, 0, 0, 0};
#if defined(__clang_major__) && (__clang_major__ >= 23)
    i32x8 z8 = {0, 0, 0, 0, 0, 0, 0, 0};
    __builtin_amdgcn_tensor_load_to_lds(g0, g1, z4, z4, z8, 0);
#else
    __builtin_amdgcn_tensor_load_to_lds(g0, g1, z4, z4, 0);
#endif
}
#endif

// --------------------------- packing kernels -------------------------------

// emb (V,300) f32 -> embH (V,320) f16, zero padded
__global__ __launch_bounds__(256) void k_emb_pack(const float* __restrict__ emb,
                                                  _Float16* __restrict__ embH) {
    int idx = blockIdx.x * 256 + threadIdx.x;
    if (idx >= V_ * EP_) return;
    int e = idx % EP_, v = idx / EP_;
    float x = (e < E_) ? emb[(size_t)v * E_ + e] : 0.0f;
    embH[idx] = (_Float16)x;
}

// W (4H, Ksrc) f32 -> packed f16 B-fragments: dst[((nT*KT+kt)*32+l)*16+h]
__global__ __launch_bounds__(256) void k_pack_w(const float* __restrict__ W,
                                                _Float16* __restrict__ dst,
                                                int KT, int Ksrc) {
    int idx = blockIdx.x * 256 + threadIdx.x;
    int total = 64 * KT * 512;
    if (idx >= total) return;
    int h  = idx & 15;
    int l  = (idx >> 4) & 31;
    int kt = (idx >> 9) % KT;
    int nT = (idx >> 9) / KT;
    int n  = nT * 16 + (l & 15);
    int k  = kt * 32 + kOfHalf(l, h);
    float x = (k < Ksrc) ? W[(size_t)n * Ksrc + k] : 0.0f;
    dst[idx] = (_Float16)x;
}

__global__ __launch_bounds__(256) void k_bsum(const float* __restrict__ bih,
                                              const float* __restrict__ bhh,
                                              float* __restrict__ bs) {
    int i = blockIdx.x * 256 + threadIdx.x;
    if (i < G4_) bs[i] = bih[i] + bhh[i];
}

// ----------------------- input-projection GEMM -----------------------------
// xg[m,n] = sum_k embH[row(m),k] * Wih[n,k] + bsum[n] ;  m = t*32+b
// Output stored in per-lane fragment order: [t][mT][g][w][lane][r] (halves),
// so the scan consumes one 16B LDS/DS vector per output tile.
__global__ __launch_bounds__(256) void k_xg_gemm(
    const _Float16* __restrict__ embH,
    const int* __restrict__ inputs, const int* __restrict__ lens,
    const _Float16* __restrict__ WihPf, const _Float16* __restrict__ WihPb,
    const float* __restrict__ bsumF, const float* __restrict__ bsumB,
    _Float16* __restrict__ xgF, _Float16* __restrict__ xgB) {

    const int dir = blockIdx.y;
    const _Float16* Wp = dir ? WihPb : WihPf;
    const float*    bs = dir ? bsumB : bsumF;
    _Float16*       xg = dir ? xgB : xgF;

    int tid  = threadIdx.x;
    int lane = tid & 31;
    int gw   = blockIdx.x * 8 + (tid >> 5);   // global wave id (16384 total)
    int mTile = gw >> 4;                      // 0..1023  (= t*2 + mT)
    int nG    = gw & 15;                      // 0..15 (4 N-tiles each)

    int m  = mTile * 16 + (lane & 15);
    int t  = m >> 5, b = m & 31;
    int len = lens[b];
    int tt  = dir ? ((t < len) ? (len - 1 - t) : t) : t;
    int row = inputs[b * T_ + tt];

    int hi = lane >> 4;
    int nc = lane & 15;
    const _Float16* arow = embH + (size_t)row * EP_;

    v8f acc[4];
#pragma unroll
    for (int j = 0; j < 4; ++j) {
        float bv = bs[(nG * 4 + j) * 16 + nc];
#pragma unroll
        for (int r = 0; r < 8; ++r) acc[j][r] = bv;
    }

    for (int kT = 0; kT < KT_IH; ++kT) {
        V16H a;
        a.q[0] = *(const uint4*)(arow + kT * 32 + hi * 8);
        a.q[1] = *(const uint4*)(arow + kT * 32 + hi * 8 + 16);
#pragma unroll
        for (int j = 0; j < 4; ++j) {
            int nT = nG * 4 + j;
            v16h bf = *(const v16h*)(Wp + ((size_t)(nT * KT_IH + kT) * 32 + lane) * 16);
            acc[j] = wmma_f16(a.v, bf, acc[j]);
        }
    }

#pragma unroll
    for (int j = 0; j < 4; ++j) {
        int nT = nG * 4 + j;
        int g  = nT >> 4;       // gate 0..3
        int ww = nT & 15;       // h-column block 0..15
        V8H p;
#pragma unroll
        for (int r = 0; r < 8; ++r) p.h[r] = (_Float16)acc[j][r];
        size_t base = ((((size_t)mTile * 4 + g) * 16 + ww) * 32 + lane) * 8;
        *(uint4*)(xg + base) = p.q;
    }
}

// ----------------------------- recurrent scan ------------------------------
// One workgroup per direction, 16 waves. Wave w owns h-columns [w*16,w*16+16)
// and the 4 gate tiles (i,f,g,o) for both M tiles. h double-buffered in LDS;
// xg slices double-buffered in LDS, prefetched one step ahead by the TDM.
__global__ __launch_bounds__(512) void k_scan(
    const _Float16* __restrict__ xgF, const _Float16* __restrict__ xgB,
    const _Float16* __restrict__ WhhPf, const _Float16* __restrict__ WhhPb,
    float* __restrict__ histF, float* __restrict__ histB) {

    const int dir = blockIdx.x;
    const _Float16* xg = dir ? xgB : xgF;
    const _Float16* Wp = dir ? WhhPb : WhhPf;
    float* hist = dir ? histB : histF;

    __shared__ __align__(16) _Float16 hbuf[2][B_][264];   // 264 = 256 + 8 pad
    extern __shared__ __align__(16) _Float16 xbuf[];      // 2 * XSLICE_ halves

    int tid  = threadIdx.x;
    int lane = tid & 31;
    int w    = tid >> 5;
    int hi   = lane >> 4;
    int nc   = lane & 15;
    int hcol = w * 16 + nc;

    // zero h buffers; kick off prefetch of xg slice for t=0 into xbuf[0]
#if USE_TDM
    if (w == 0)
        tdm_load_1d((unsigned)(uintptr_t)(xbuf), xg, XSLICE_);
#endif
    for (int i = tid; i < 2 * B_ * 264; i += 512)
        ((_Float16*)hbuf)[i] = (_Float16)0.0f;
#if USE_TDM
    if (w == 0) __builtin_amdgcn_s_wait_tensorcnt(0);
#else
    for (int i = tid; i < XSLICE_ / 8; i += 512)
        *(uint4*)(xbuf + (size_t)i * 8) = *(const uint4*)(xg + (size_t)i * 8);
#endif
    __syncthreads();

    float cst[2][8];
#pragma unroll
    for (int mT = 0; mT < 2; ++mT)
#pragma unroll
        for (int r = 0; r < 8; ++r) cst[mT][r] = 0.0f;

    for (int t = 0; t < T_; ++t) {
        int cur = t & 1, nxt = cur ^ 1;

        // prefetch next step's xg slice into the other LDS buffer
        if (t + 1 < T_) {
#if USE_TDM
            if (w == 0)
                tdm_load_1d((unsigned)(uintptr_t)(xbuf + (size_t)nxt * XSLICE_),
                            xg + (size_t)(t + 1) * XSLICE_, XSLICE_);
#else
            for (int i = tid; i < XSLICE_ / 8; i += 512)
                *(uint4*)(xbuf + (size_t)nxt * XSLICE_ + (size_t)i * 8) =
                    *(const uint4*)(xg + (size_t)(t + 1) * XSLICE_ + (size_t)i * 8);
#endif
        }

        // accumulators = xg tile (one ds_load_b128 per tile)
        const _Float16* xb = xbuf + (size_t)cur * XSLICE_;
        v8f acc[2][4];
#pragma unroll
        for (int mT = 0; mT < 2; ++mT)
#pragma unroll
            for (int g = 0; g < 4; ++g) {
                V8H p;
                p.q = *(const uint4*)(xb + (((mT * 4 + g) * 16 + w) * 32 + lane) * 8);
#pragma unroll
                for (int r = 0; r < 8; ++r) acc[mT][g][r] = (float)p.h[r];
            }

        for (int kT = 0; kT < KT_HH; ++kT) {
            V16H a[2];
#pragma unroll
            for (int mT = 0; mT < 2; ++mT) {
                int mrow  = mT * 16 + nc;
                int kbase = kT * 32 + hi * 8;
                a[mT].q[0] = *(const uint4*)&hbuf[cur][mrow][kbase];
                a[mT].q[1] = *(const uint4*)&hbuf[cur][mrow][kbase + 16];
            }
#pragma unroll
            for (int g = 0; g < 4; ++g) {
                int nT = g * 16 + w;
                v16h bf = *(const v16h*)(Wp + ((size_t)(nT * KT_HH + kT) * 32 + lane) * 16);
                acc[0][g] = wmma_f16(a[0].v, bf, acc[0][g]);
                acc[1][g] = wmma_f16(a[1].v, bf, acc[1][g]);
            }
        }

#pragma unroll
        for (int mT = 0; mT < 2; ++mT) {
            float h8[8];
#pragma unroll
            for (int r = 0; r < 8; ++r) {
                float iv = sigmoidf_(acc[mT][0][r]);
                float fv = sigmoidf_(acc[mT][1][r]);
                float gv = tanhf(acc[mT][2][r]);
                float ov = sigmoidf_(acc[mT][3][r]);
                float c  = fv * cst[mT][r] + iv * gv;
                cst[mT][r] = c;
                float h = ov * tanhf(c);
                h8[r] = h;
                int ml = mT * 16 + hi * 8 + r;
                hbuf[nxt][ml][hcol] = (_Float16)h;
            }
            // hist in fragment layout: [t][mT][w][lane][r]
            float* hp = hist + ((((size_t)t * 2 + mT) * 16 + w) * 32 + lane) * 8;
            float4 v0 = {h8[0], h8[1], h8[2], h8[3]};
            float4 v1 = {h8[4], h8[5], h8[6], h8[7]};
            *(float4*)(hp)     = v0;
            *(float4*)(hp + 4) = v1;
        }

#if USE_TDM
        if (w == 0) __builtin_amdgcn_s_wait_tensorcnt(0);
#endif
        __syncthreads();
    }
}

// --------------------------- mask index + gather ---------------------------

__global__ __launch_bounds__(64) void k_maskidx(const unsigned char* __restrict__ fmask,
                                                const unsigned char* __restrict__ bmask,
                                                int* __restrict__ fidx,
                                                int* __restrict__ bidx) {
    int tid = threadIdx.x;
    if (tid >= 64) return;
    const unsigned char* mrow = (tid < 32 ? fmask : bmask) + (size_t)(tid & 31) * T_;
    int* orow = (tid < 32 ? fidx : bidx) + (tid & 31) * OUT_;
    int c = 0;
    for (int t = 0; t < T_ && c < OUT_; ++t)
        if (mrow[t]) orow[c++] = t;
}

// hist fragment index for value (t, batch-row ml, h-column hcol)
__device__ __forceinline__ size_t hist_idx(int t, int ml, int hcol) {
    int mT = ml >> 4, rem = ml & 15, hi = rem >> 3, r = rem & 7;
    int w = hcol >> 4, nc = hcol & 15, lane = hi * 16 + nc;
    return ((((size_t)t * 2 + mT) * 16 + w) * 32 + lane) * 8 + r;
}

__global__ __launch_bounds__(256) void k_out(const float* __restrict__ histF,
                                             const float* __restrict__ histB,
                                             const int* __restrict__ fidx,
                                             const int* __restrict__ bidx,
                                             float* __restrict__ out) {
    int idx = blockIdx.x * 256 + threadIdx.x;     // (b*256+j)*512+ch
    int ch = idx & 511;
    int j  = (idx >> 9) & 255;
    int b  = idx >> 17;
    float v;
    if (ch < H_) {
        int t = fidx[b * OUT_ + j];
        v = histF[hist_idx(t, b, ch)];
    } else {
        int t = bidx[b * OUT_ + (OUT_ - 1 - j)];  // k == OUT always (len >= OUT)
        v = histB[hist_idx(t, b, ch - H_)];
    }
    out[idx] = v;
}

// ------------------------------- launcher ----------------------------------

extern "C" void kernel_launch(void* const* d_in, const int* in_sizes, int n_in,
                              void* d_out, int out_size, void* d_ws, size_t ws_size,
                              hipStream_t stream) {
    const int*   inputs = (const int*)d_in[0];
    const int*   lens   = (const int*)d_in[1];
    const unsigned char* fmask = (const unsigned char*)d_in[2];
    const unsigned char* bmask = (const unsigned char*)d_in[3];
    // d_in[4] = out_seq_length (compile-time OUT_)
    const float* emb   = (const float*)d_in[5];
    const float* Wih_f = (const float*)d_in[6];
    const float* Whh_f = (const float*)d_in[7];
    const float* bih_f = (const float*)d_in[8];
    const float* bhh_f = (const float*)d_in[9];
    const float* Wih_b = (const float*)d_in[10];
    const float* Whh_b = (const float*)d_in[11];
    const float* bih_b = (const float*)d_in[12];
    const float* bhh_b = (const float*)d_in[13];
    float* out = (float*)d_out;

    char* ws = (char*)d_ws;
    size_t off = 0;
    auto alloc = [&](size_t bytes) -> char* {
        char* p = ws + off;
        off = (off + bytes + 255) & ~(size_t)255;
        return p;
    };

    _Float16* embH  = (_Float16*)alloc((size_t)V_ * EP_ * 2);
    _Float16* WihPf = (_Float16*)alloc((size_t)64 * KT_IH * 512 * 2);
    _Float16* WihPb = (_Float16*)alloc((size_t)64 * KT_IH * 512 * 2);
    _Float16* WhhPf = (_Float16*)alloc((size_t)64 * KT_HH * 512 * 2);
    _Float16* WhhPb = (_Float16*)alloc((size_t)64 * KT_HH * 512 * 2);
    float*    bsumF = (float*)alloc(G4_ * 4);
    float*    bsumB = (float*)alloc(G4_ * 4);
    _Float16* xgF   = (_Float16*)alloc((size_t)T_ * B_ * G4_ * 2);
    _Float16* xgB   = (_Float16*)alloc((size_t)T_ * B_ * G4_ * 2);
    float*    histF = (float*)alloc((size_t)T_ * B_ * H_ * 4);
    float*    histB = (float*)alloc((size_t)T_ * B_ * H_ * 4);
    int*      fidx  = (int*)alloc(B_ * OUT_ * 4);
    int*      bidx  = (int*)alloc(B_ * OUT_ * 4);

    // 1) convert / repack parameters into WMMA fragment layouts
    k_emb_pack<<<(V_ * EP_ + 255) / 256, 256, 0, stream>>>(emb, embH);
    k_pack_w<<<(64 * KT_IH * 512 + 255) / 256, 256, 0, stream>>>(Wih_f, WihPf, KT_IH, E_);
    k_pack_w<<<(64 * KT_IH * 512 + 255) / 256, 256, 0, stream>>>(Wih_b, WihPb, KT_IH, E_);
    k_pack_w<<<(64 * KT_HH * 512 + 255) / 256, 256, 0, stream>>>(Whh_f, WhhPf, KT_HH, H_);
    k_pack_w<<<(64 * KT_HH * 512 + 255) / 256, 256, 0, stream>>>(Whh_b, WhhPb, KT_HH, H_);
    k_bsum<<<4, 256, 0, stream>>>(bih_f, bhh_f, bsumF);
    k_bsum<<<4, 256, 0, stream>>>(bih_b, bhh_b, bsumB);

    // 2) input projection GEMM, both directions (16384 waves x 2)
    k_xg_gemm<<<dim3(2048, 2), 256, 0, stream>>>(embH, inputs, lens,
                                                 WihPf, WihPb, bsumF, bsumB, xgF, xgB);

    // 3) recurrent scan, one persistent workgroup per direction (TDM prefetch)
    k_scan<<<2, 512, XBUF_BYTES, stream>>>(xgF, xgB, WhhPf, WhhPb, histF, histB);

    // 4) mask indices + gather into (B, OUT, 2H) output
    k_maskidx<<<1, 64, 0, stream>>>(fmask, bmask, fidx, bidx);
    k_out<<<(B_ * OUT_ * 2 * H_) / 256, 256, 0, stream>>>(histF, histB, fidx, bidx, out);

    (void)in_sizes; (void)n_in; (void)out_size; (void)ws_size;
}
